// AttentionCapModule_18433999634627
// MI455X (gfx1250) — matrix operand order repaired and checked by reading, not compiled, not run on hardware
//
#include <hip/hip_runtime.h>
#include <hip/hip_bf16.h>

#define F_DIM 128
#define EMB   300
#define EMBP  320
#define HID   512
#define NOBJ  4096
#define KNBR  64
#define CIN1  940     // HID + F_DIM + EMB
#define CIN1P 960
#define CIN2  640     // F_DIM + HID
#define G3    1536    // 3*HID

typedef __attribute__((ext_vector_type(16))) __bf16 v16bf;
typedef __attribute__((ext_vector_type(8)))  float  v8f;

union Frag16 { uint4 u[2]; v16bf v; };

// Async DMA: global -> LDS, 16 bytes per lane, tracked by ASYNCcnt.
__device__ __forceinline__ void async_g2l_b128(void* lds_dst, const void* gsrc)
{
    unsigned lds_off = (unsigned)(uintptr_t)lds_dst;   // low 32b = LDS offset
    asm volatile("global_load_async_to_lds_b128 %0, %1, off"
                 :: "v"(lds_off), "v"(gsrc) : "memory");
}
__device__ __forceinline__ void wait_async0()
{
    asm volatile("s_wait_asynccnt 0x0" ::: "memory");
}

// y[0..rows) = W @ x + bias, via v_wmma_f32_16x16x32_bf16.
// W is packed fragment-major: per (M-tile, K-chunk) a contiguous 1KB block of
// all 32 lanes' A fragments. x is a bf16 LDS vector (B broadcast into all 16
// columns; column 0 of D extracted by lanes 0/16). Two M-tiles per iteration
// share one B fragment; a 1-deep software pipeline keeps the next chunk's
// A/B loads in flight across the current WMMA (partial waits instead of
// full drains).
__device__ __forceinline__
void matvec_wmma(const __bf16* __restrict__ W, int Kp, int rows,
                 const __bf16* __restrict__ xb,
                 float* __restrict__ y, __bf16* __restrict__ yb,
                 const float* __restrict__ bias,
                 int wave, int lane, int nwaves)
{
    const int rowTiles = (rows + 15) >> 4;
    const int kTiles   = Kp >> 5;
    const int selB     = (lane < 16) ? 0 : 8;

    auto writeback = [&](int mt, const v8f& c) {
        if (lane == 0 || lane == 16) {
            const int base = mt * 16 + ((lane == 16) ? 8 : 0);
#pragma unroll
            for (int r = 0; r < 8; ++r) {
                int row = base + r;
                if (row < rows) {
                    float val = c[r] + (bias ? bias[row] : 0.f);
                    if (y)  y[row]  = val;
                    if (yb) yb[row] = (__bf16)val;
                }
            }
        }
    };

    for (int mt0 = wave; mt0 < rowTiles; mt0 += 2 * nwaves) {
        const int  mt1 = mt0 + nwaves;
        const bool two = (mt1 < rowTiles);              // wave-uniform
        const __bf16* w0 = W + (size_t)mt0 * kTiles * 512 + lane * 16;
        const __bf16* w1 = W + (size_t)mt1 * kTiles * 512 + lane * 16;
        v8f c0 = {}, c1 = {};

        Frag16 bN, a0N, a1N = {};
        bN.u[0]  = *reinterpret_cast<const uint4*>(xb + selB);
        bN.u[1]  = *reinterpret_cast<const uint4*>(xb + selB + 16);
        a0N.u[0] = *reinterpret_cast<const uint4*>(w0);
        a0N.u[1] = *reinterpret_cast<const uint4*>(w0 + 8);
        if (two) {
            a1N.u[0] = *reinterpret_cast<const uint4*>(w1);
            a1N.u[1] = *reinterpret_cast<const uint4*>(w1 + 8);
        }
        for (int kc = 0; kc < kTiles; ++kc) {
            Frag16 b = bN, a0 = a0N, a1 = a1N;
            if (kc + 1 < kTiles) {                      // issue next chunk early
                const __bf16* xp = xb + (kc + 1) * 32 + selB;
                bN.u[0]  = *reinterpret_cast<const uint4*>(xp);
                bN.u[1]  = *reinterpret_cast<const uint4*>(xp + 16);
                a0N.u[0] = *reinterpret_cast<const uint4*>(w0 + (size_t)(kc + 1) * 512);
                a0N.u[1] = *reinterpret_cast<const uint4*>(w0 + (size_t)(kc + 1) * 512 + 8);
                if (two) {
                    a1N.u[0] = *reinterpret_cast<const uint4*>(w1 + (size_t)(kc + 1) * 512);
                    a1N.u[1] = *reinterpret_cast<const uint4*>(w1 + (size_t)(kc + 1) * 512 + 8);
                }
            }
            c0 = __builtin_amdgcn_wmma_f32_16x16x32_bf16(
                     false, a0.v, false, b.v, (short)0, c0, false, false);
            if (two)
                c1 = __builtin_amdgcn_wmma_f32_16x16x32_bf16(
                         false, a1.v, false, b.v, (short)0, c1, false, false);
        }
        writeback(mt0, c0);
        if (two) writeback(mt1, c1);
    }
}

__device__ __forceinline__ float sigmoidf_(float x) { return 1.f / (1.f + __expf(-x)); }

// ---- prep kernels -----------------------------------------------------------

// Repack f32 row-major [rows x cols] into bf16 fragment-major [rowsP x Kp].
__global__ void pack_frag(const float* __restrict__ src, __bf16* __restrict__ dst,
                          int rows, int cols, int rowsP, int Kp)
{
    size_t idx = (size_t)blockIdx.x * blockDim.x + threadIdx.x;
    size_t total = (size_t)rowsP * Kp;
    if (idx >= total) return;
    int    j     = (int)(idx & 15);
    int    lane  = (int)((idx >> 4) & 31);
    size_t blk   = idx >> 9;                       // 512 elements per block
    int    kT    = Kp >> 5;
    int    mt    = (int)(blk / kT);
    int    kc    = (int)(blk - (size_t)mt * kT);
    int    selA  = (lane < 16) ? 0 : 8;
    int    row   = mt * 16 + (lane & 15);
    int    col   = kc * 32 + selA + ((j < 8) ? j : (8 + j));
    float  v     = (row < rows && col < cols) ? src[(size_t)row * cols + col] : 0.f;
    dst[idx] = (__bf16)v;
}

// u = W_v^T @ W_a[0]   (scores_k = V_r[k] . u ; W_h term is softmax-invariant)
__global__ void compute_u(const float* __restrict__ Wv, const float* __restrict__ Wa,
                          float* __restrict__ u)
{
    int j = threadIdx.x;
    if (j < F_DIM) {
        float s = 0.f;
        for (int i = 0; i < F_DIM; ++i) s += Wa[i] * Wv[i * F_DIM + j];
        u[j] = s;
    }
}

// ---- persistent scan kernel: one workgroup, 4096 sequential steps -----------

__global__ __launch_bounds__(1024)
void cap_persistent(const float* __restrict__ V, const float* __restrict__ E,
                    const __bf16* __restrict__ WE, const __bf16* __restrict__ WFC1,
                    const __bf16* __restrict__ WIHF, const __bf16* __restrict__ WHHF,
                    const __bf16* __restrict__ WIHL, const __bf16* __restrict__ WHHL,
                    const __bf16* __restrict__ WFC2, const __bf16* __restrict__ WFC3,
                    const float* __restrict__ b_fc1, const float* __restrict__ b_ih_f,
                    const float* __restrict__ b_hh_f, const float* __restrict__ b_fc2,
                    const float* __restrict__ b_ih_l, const float* __restrict__ b_hh_l,
                    const float* __restrict__ b_fc3, const float* __restrict__ g_u,
                    float* __restrict__ out)
{
    __shared__ float s_token[EMBP];                 // f32 token (for output)
    __shared__ float s_h1[HID], s_h2[HID];          // f32 recurrent carries
    __shared__ float s_gi[G3], s_gh[G3];            // GRU gate pre-activations
    __shared__ __align__(16) float s_E[KNBR * F_DIM];
    __shared__ float s_v[F_DIM], s_u[F_DIM], s_sc[KNBR];
    // bf16 matvec inputs, B-fragment-ready
    __shared__ __align__(16) __bf16 s_xb[CIN1P];    // [h2 | v | x_t | 0pad]
    __shared__ __align__(16) __bf16 s_gb[HID];      // g1 / g2
    __shared__ __align__(16) __bf16 s_h1b[HID];
    __shared__ __align__(16) __bf16 s_h2b[HID];
    __shared__ __align__(16) __bf16 s_fusedb[CIN2]; // [aggr | h1]
    __shared__ __align__(16) __bf16 s_tokenb[EMBP];

    const int tid  = threadIdx.x;
    const int lane = tid & 31;
    const int wave = tid >> 5;

    if (tid < EMBP) s_tokenb[tid] = (__bf16)0.f;
    if (tid < HID) { s_h1[tid] = 0.f; s_h2[tid] = 0.f;
                     s_h1b[tid] = (__bf16)0.f; s_h2b[tid] = (__bf16)0.f; }
    if (tid >= CIN1 && tid < CIN1P) s_xb[tid] = (__bf16)0.f;   // permanent pad
    if (tid < F_DIM) s_u[tid] = g_u[tid];
    __syncthreads();

    // stage step 0: async E tile DMA + v load
    async_g2l_b128(&s_E[tid * 8],     E + tid * 8);
    async_g2l_b128(&s_E[tid * 8 + 4], E + tid * 8 + 4);
    if (tid < F_DIM) s_v[tid] = V[tid];

    for (int t = 0; t < NOBJ; ++t) {
        // phase 1: complete staged E/v
        wait_async0();
        __syncthreads();

        // phase 2: inp = [h2, v, W_e@token] (bf16); attention scores (V_r . u)
        if (tid < HID)   s_xb[tid] = s_h2b[tid];
        if (tid < F_DIM) s_xb[HID + tid] = (__bf16)s_v[tid];
        if (tid < KNBR) {
            float s = 0.f;
            const float* e = &s_E[tid * F_DIM];
            for (int j = 0; j < F_DIM; ++j) s += (s_v[j] + e[j]) * s_u[j];
            s_sc[tid] = s;
        }
        matvec_wmma(WE, EMBP, EMB, s_tokenb, nullptr, s_xb + HID + F_DIM,
                    nullptr, wave, lane, 32);
        __syncthreads();

        // phase 3: g1 = W_fc1 @ inp + b_fc1 ; softmax over 64 scores
        if (tid == 0) {
            float m = s_sc[0];
            for (int k = 1; k < KNBR; ++k) m = fmaxf(m, s_sc[k]);
            float s = 0.f;
            for (int k = 0; k < KNBR; ++k) { float ex = __expf(s_sc[k] - m); s_sc[k] = ex; s += ex; }
            float inv = 1.f / s;
            for (int k = 0; k < KNBR; ++k) s_sc[k] *= inv;
        }
        matvec_wmma(WFC1, CIN1P, HID, s_xb, nullptr, s_gb, b_fc1, wave, lane, 32);
        __syncthreads();

        // phase 4: aggr = attn @ V_r (last reader of s_E/s_v); GRU-f gates
        if (tid < F_DIM) {
            float a = 0.f;
            for (int k = 0; k < KNBR; ++k) a += s_sc[k] * (s_v[tid] + s_E[k * F_DIM + tid]);
            s_fusedb[tid] = (__bf16)a;
        }
        if (wave < 16) matvec_wmma(WIHF, HID, G3, s_gb,  s_gi, nullptr, b_ih_f, wave,      lane, 16);
        else           matvec_wmma(WHHF, HID, G3, s_h1b, s_gh, nullptr, b_hh_f, wave - 16, lane, 16);
        __syncthreads();

        // phase 5: GRU-f elementwise; stage next step's E (async DMA) and v —
        // overlaps phases 6-9 compute; completed at next loop head.
        if (tid < HID) {
            float r = sigmoidf_(s_gi[tid] + s_gh[tid]);
            float z = sigmoidf_(s_gi[HID + tid] + s_gh[HID + tid]);
            float n = tanhf(s_gi[2 * HID + tid] + r * s_gh[2 * HID + tid]);
            float h = (1.f - z) * n + z * s_h1[tid];
            s_h1[tid] = h;
            s_h1b[tid] = (__bf16)h;
            s_fusedb[F_DIM + tid] = (__bf16)h;
        }
        if (t + 1 < NOBJ) {
            const float* g = E + (size_t)(t + 1) * (KNBR * F_DIM) + tid * 8;
            async_g2l_b128(&s_E[tid * 8],     g);
            async_g2l_b128(&s_E[tid * 8 + 4], g + 4);
            if (tid < F_DIM) s_v[tid] = V[(size_t)(t + 1) * F_DIM + tid];
        }
        if (t + 2 < NOBJ && tid < 256)
            __builtin_prefetch(E + (size_t)(t + 2) * (KNBR * F_DIM) + tid * 32, 0, 3);
        __syncthreads();

        // phase 6: g2 = W_fc2 @ fused + b_fc2
        matvec_wmma(WFC2, CIN2, HID, s_fusedb, nullptr, s_gb, b_fc2, wave, lane, 32);
        __syncthreads();

        // phase 7: GRU-l gate matvecs
        if (wave < 16) matvec_wmma(WIHL, HID, G3, s_gb,  s_gi, nullptr, b_ih_l, wave,      lane, 16);
        else           matvec_wmma(WHHL, HID, G3, s_h2b, s_gh, nullptr, b_hh_l, wave - 16, lane, 16);
        __syncthreads();

        // phase 8: GRU-l elementwise; update h2
        if (tid < HID) {
            float r = sigmoidf_(s_gi[tid] + s_gh[tid]);
            float z = sigmoidf_(s_gi[HID + tid] + s_gh[HID + tid]);
            float n = tanhf(s_gi[2 * HID + tid] + r * s_gh[2 * HID + tid]);
            float h = (1.f - z) * n + z * s_h2[tid];
            s_h2[tid] = h;
            s_h2b[tid] = (__bf16)h;
        }
        __syncthreads();

        // phase 9: token = W_fc3 @ h2 + b_fc3  (f32 for output, bf16 for next step)
        matvec_wmma(WFC3, HID, EMB, s_h2b, s_token, s_tokenb, b_fc3, wave, lane, 32);
        __syncthreads();

        // emit token
        if (tid < EMB) out[(size_t)t * EMB + tid] = s_token[tid];
    }
}

// ---- host entry -------------------------------------------------------------

extern "C" void kernel_launch(void* const* d_in, const int* in_sizes, int n_in,
                              void* d_out, int out_size, void* d_ws, size_t ws_size,
                              hipStream_t stream)
{
    (void)in_sizes; (void)n_in; (void)out_size; (void)ws_size;

    const float* V      = (const float*)d_in[0];
    const float* E      = (const float*)d_in[1];
    const float* W_e    = (const float*)d_in[2];
    const float* W_fc1  = (const float*)d_in[3];
    const float* b_fc1  = (const float*)d_in[4];
    const float* w_ih_f = (const float*)d_in[5];
    const float* w_hh_f = (const float*)d_in[6];
    const float* b_ih_f = (const float*)d_in[7];
    const float* b_hh_f = (const float*)d_in[8];
    const float* W_v    = (const float*)d_in[9];
    /* d_in[10] = W_h: unused — softmax-invariant contribution. */
    const float* W_a    = (const float*)d_in[11];
    const float* W_fc2  = (const float*)d_in[12];
    const float* b_fc2  = (const float*)d_in[13];
    const float* w_ih_l = (const float*)d_in[14];
    const float* w_hh_l = (const float*)d_in[15];
    const float* b_ih_l = (const float*)d_in[16];
    const float* b_hh_l = (const float*)d_in[17];
    const float* W_fc3  = (const float*)d_in[18];
    const float* b_fc3  = (const float*)d_in[19];

    // bf16 fragment-major weights in workspace (~8.5 MB)
    __bf16* WE    = (__bf16*)d_ws;
    __bf16* WFC1  = WE    + (size_t)EMBP * EMBP;
    __bf16* WIHF  = WFC1  + (size_t)HID  * CIN1P;
    __bf16* WHHF  = WIHF  + (size_t)G3   * HID;
    __bf16* WIHL  = WHHF  + (size_t)G3   * HID;
    __bf16* WHHL  = WIHL  + (size_t)G3   * HID;
    __bf16* WFC2p = WHHL  + (size_t)G3   * HID;
    __bf16* WFC3p = WFC2p + (size_t)HID  * CIN2;
    float*  U     = (float*)(WFC3p + (size_t)EMBP * HID);

#define PACK(src, dst, r, c, rp, kp)                                            \
    pack_frag<<<dim3((unsigned)(((size_t)(rp) * (kp) + 255) / 256)),            \
                dim3(256), 0, stream>>>(src, dst, r, c, rp, kp)

    PACK(W_e,    WE,    EMB, EMB,  EMBP, EMBP);
    PACK(W_fc1,  WFC1,  HID, CIN1, HID,  CIN1P);
    PACK(w_ih_f, WIHF,  G3,  HID,  G3,   HID);
    PACK(w_hh_f, WHHF,  G3,  HID,  G3,   HID);
    PACK(w_ih_l, WIHL,  G3,  HID,  G3,   HID);
    PACK(w_hh_l, WHHL,  G3,  HID,  G3,   HID);
    PACK(W_fc2,  WFC2p, HID, CIN2, HID,  CIN2);
    PACK(W_fc3,  WFC3p, EMB, HID,  EMBP, HID);
#undef PACK
    compute_u<<<1, 128, 0, stream>>>(W_v, W_a, U);

    cap_persistent<<<1, 1024, 0, stream>>>(
        V, E, WE, WFC1, WIHF, WHHF, WIHL, WHHL, WFC2p, WFC3p,
        b_fc1, b_ih_f, b_hh_f, b_fc2, b_ih_l, b_hh_l, b_fc3, U,
        (float*)d_out);
}
